// SparseFormerHead_33689723470046
// MI455X (gfx1250) — matrix operand order repaired and checked by reading, not compiled, not run on hardware
//
#include <hip/hip_runtime.h>
#include <hip/hip_bf16.h>

typedef __attribute__((ext_vector_type(16))) _Float16 v16h;
typedef __attribute__((ext_vector_type(8)))  _Float16 h8;
typedef __attribute__((ext_vector_type(8)))  float    v8f;

#define HH     360
#define WW     360
#define HWSZ   129600
#define NCLS   10
#define CCH    128
#define NPROP  200
#define BB     2
#define NROWS  80000
#define NTILES 5000          /* NROWS / 16 */
#define GEMM_BLOCKS 250      /* 250 blocks * 4 waves = 1000 waves; 5000/1000 = 5 tiles each (exact) */
#define CAP    65536

/* output float offsets */
#define OFF_QF  0
#define OFF_POS 51200
#define OFF_QHS 52000
#define OFF_CLS 56000

/* workspace byte offsets */
#define WS_HEAT   ((size_t)0)            /* B*10*HW*4 = 10,368,000 */
#define WS_HEATS  ((size_t)10368000)     /* suppressed heat, same size */
#define WS_P2R    ((size_t)20736000)     /* B*HW int32 = 1,036,800 */
#define WS_HIST   ((size_t)21772800)     /* B*65536 u32 = 524,288 */
#define WS_THR    ((size_t)22297088)     /* B ints */
#define WS_CCNT   ((size_t)22297600)     /* B u32 */
#define WS_SELF   ((size_t)22298112)     /* B*NPROP int32 = 1600 */
#define WS_CAND   ((size_t)22300672)     /* B*CAP*8 = 1,048,576 */

/* ---------------- pos -> sparse row map ---------------- */
__global__ void build_map_kernel(const int* __restrict__ ind, int* __restrict__ p2r) {
  int i = blockIdx.x * blockDim.x + threadIdx.x;
  if (i >= NROWS) return;
  int b = ind[3 * i + 0];
  int y = ind[3 * i + 1];
  int x = ind[3 * i + 2];
  p2r[(size_t)b * HWSZ + y * WW + x] = i;
}

/* ---------------- WMMA head GEMM + sigmoid scatter ---------------- */
__global__ __launch_bounds__(128) void head_gemm_kernel(
    const float* __restrict__ F, const float* __restrict__ W1,
    const float* __restrict__ b1, const float* __restrict__ W2,
    const float* __restrict__ b2, const int* __restrict__ ind,
    float* __restrict__ heat) {
  __shared__ __align__(32) _Float16 Bf1[4 * 8 * 32 * 16]; /* [kk][nt][lane][16] */
  __shared__ __align__(32) _Float16 Bf2[4 * 32 * 16];     /* [kk][lane][16]     */
  __shared__ __align__(32) _Float16 A2s[4][16 * CCH];     /* per-wave relu tile */

  const int tid = threadIdx.x;

  /* one-time: build fragment-ready f16 copies of W1 and (padded) W2 */
  for (int i = tid; i < 4 * 8 * 32 * 16; i += 128) {
    int h = i & 15, lane = (i >> 4) & 31, nt = (i >> 9) & 7, kk = i >> 12;
    int krow = 32 * kk + 8 * (lane >> 4) + h + 8 * (h >> 3);
    int col = nt * 16 + (lane & 15);
    Bf1[i] = (_Float16)W1[krow * CCH + col];
  }
  for (int i = tid; i < 4 * 32 * 16; i += 128) {
    int h = i & 15, lane = (i >> 4) & 31, kk = i >> 9;
    int krow = 32 * kk + 8 * (lane >> 4) + h + 8 * (h >> 3);
    int col = lane & 15;
    Bf2[i] = (col < NCLS) ? (_Float16)W2[krow * NCLS + col] : (_Float16)0.0f;
  }
  __syncthreads();

  const int wave = tid >> 5;
  const int lane = tid & 31;
  const int halfid = lane >> 4;
  const int col = lane & 15;
  const int wgid = blockIdx.x * 4 + wave;
  const int wtot = gridDim.x * 4;

  for (int t = wgid; t < NTILES; t += wtot) {
    const int m0 = t * 16;
    if (t + wtot < NTILES) { /* prefetch next tile's row for this lane */
      __builtin_prefetch(F + (size_t)((t + wtot) * 16 + col) * CCH, 0, 3);
    }

    v8f acc[8] = {};
    const float* Fr = F + (size_t)(m0 + col) * CCH;
#pragma unroll
    for (int kk = 0; kk < 4; ++kk) {
      const int run1 = 32 * kk + 8 * halfid;
      float4 a0 = *(const float4*)(Fr + run1);
      float4 a1 = *(const float4*)(Fr + run1 + 4);
      float4 a2 = *(const float4*)(Fr + run1 + 16);
      float4 a3 = *(const float4*)(Fr + run1 + 20);
      v16h a;
      a[0] = (_Float16)a0.x;  a[1] = (_Float16)a0.y;  a[2] = (_Float16)a0.z;  a[3] = (_Float16)a0.w;
      a[4] = (_Float16)a1.x;  a[5] = (_Float16)a1.y;  a[6] = (_Float16)a1.z;  a[7] = (_Float16)a1.w;
      a[8] = (_Float16)a2.x;  a[9] = (_Float16)a2.y;  a[10] = (_Float16)a2.z; a[11] = (_Float16)a2.w;
      a[12] = (_Float16)a3.x; a[13] = (_Float16)a3.y; a[14] = (_Float16)a3.z; a[15] = (_Float16)a3.w;
#pragma unroll
      for (int nt = 0; nt < 8; ++nt) {
        const v16h bfr = *(const v16h*)&Bf1[((kk * 8 + nt) * 32 + lane) * 16];
        acc[nt] = __builtin_amdgcn_wmma_f32_16x16x32_f16(false, a, false, bfr,
                                                         (short)0, acc[nt], false, false);
      }
    }

    /* bias + relu, stage 16x128 tile to LDS as f16 (row major) */
#pragma unroll
    for (int nt = 0; nt < 8; ++nt) {
      const float bias = b1[nt * 16 + col];
#pragma unroll
      for (int r = 0; r < 8; ++r) {
        float v = acc[nt][r] + bias;
        v = v > 0.0f ? v : 0.0f;
        A2s[wave][(r + 8 * halfid) * CCH + nt * 16 + col] = (_Float16)v;
      }
    }
    __syncthreads();

    /* GEMM2: (16x128) x (128x16-padded) */
    v8f acc2 = {};
#pragma unroll
    for (int kk = 0; kk < 4; ++kk) {
      const int run1 = 32 * kk + 8 * halfid;
      h8 lo = *(const h8*)&A2s[wave][col * CCH + run1];
      h8 hi = *(const h8*)&A2s[wave][col * CCH + run1 + 16];
      v16h a;
#pragma unroll
      for (int e = 0; e < 8; ++e) { a[e] = lo[e]; a[8 + e] = hi[e]; }
      const v16h bfr = *(const v16h*)&Bf2[(kk * 32 + lane) * 16];
      acc2 = __builtin_amdgcn_wmma_f32_16x16x32_f16(false, a, false, bfr,
                                                    (short)0, acc2, false, false);
    }

    const float bias2 = (col < NCLS) ? b2[col] : 0.0f;
#pragma unroll
    for (int r = 0; r < 8; ++r) {
      const int srow = m0 + r + 8 * halfid;
      const float v = acc2[r] + bias2;
      const float s = 1.0f / (1.0f + expf(-v));
      if (col < NCLS) {
        const int bb = ind[srow * 3 + 0];
        const int yy = ind[srow * 3 + 1];
        const int xx = ind[srow * 3 + 2];
        heat[(size_t)(bb * NCLS + col) * HWSZ + yy * WW + xx] = s;
      }
    }
    __syncthreads();
  }
}

/* ---------------- 3x3 NMS + high-16-bit histogram ---------------- */
__global__ void nms_hist_kernel(const float* __restrict__ heat,
                                float* __restrict__ heat_s,
                                unsigned* __restrict__ hist) {
  const int n = BB * NCLS * HWSZ;
  for (int i = blockIdx.x * blockDim.x + threadIdx.x; i < n;
       i += gridDim.x * blockDim.x) {
    const int b = i / (NCLS * HWSZ);
    const int rem = i - b * (NCLS * HWSZ);
    const int c = rem / HWSZ;
    const int pix = rem - c * HWSZ;
    const int y = pix / WW;
    const int x = pix - y * WW;
    const float h = heat[i];
    float out;
    if (c >= 8) {
      out = h;
    } else if (y == 0 || y == HH - 1 || x == 0 || x == WW - 1) {
      out = 0.0f;
    } else {
      const float* base = heat + (size_t)(b * NCLS + c) * HWSZ;
      float m = h;
#pragma unroll
      for (int dy = -1; dy <= 1; ++dy)
#pragma unroll
        for (int dx = -1; dx <= 1; ++dx)
          m = fmaxf(m, base[(y + dy) * WW + (x + dx)]);
      out = (h == m) ? h : 0.0f;
    }
    heat_s[i] = out;
    if (out > 0.0f)
      atomicAdd(&hist[(size_t)b * 65536 + (__float_as_uint(out) >> 16)], 1u);
  }
}

/* ---------------- find per-batch threshold bin ---------------- */
__global__ __launch_bounds__(256) void select_thr_kernel(
    const unsigned* __restrict__ hist, int* __restrict__ thr) {
  __shared__ unsigned ls[256];
  const int b = blockIdx.x;
  const int t = threadIdx.x;
  unsigned s = 0;
  for (int j = 0; j < 256; ++j) s += hist[(size_t)b * 65536 + t * 256 + j];
  ls[t] = s;
  __syncthreads();
  if (t == 0) {
    unsigned cum = 0;
    int coarse = -1;
    for (int q = 255; q >= 0; --q) {
      if (cum + ls[q] >= (unsigned)NPROP) { coarse = q; break; }
      cum += ls[q];
    }
    int thrv = 0;
    if (coarse >= 0) {
      for (int bin = coarse * 256 + 255; bin >= coarse * 256; --bin) {
        cum += hist[(size_t)b * 65536 + bin];
        if (cum >= (unsigned)NPROP) { thrv = bin; break; }
      }
    }
    thr[b] = thrv;
  }
}

/* ---------------- compact candidates >= threshold ---------------- */
__global__ void compact_kernel(const float* __restrict__ heat_s,
                               const int* __restrict__ thr,
                               unsigned* __restrict__ ccnt,
                               unsigned long long* __restrict__ cand) {
  const int n = BB * NCLS * HWSZ;
  for (int i = blockIdx.x * blockDim.x + threadIdx.x; i < n;
       i += gridDim.x * blockDim.x) {
    const float v = heat_s[i];
    if (v > 0.0f) {
      const int b = i / (NCLS * HWSZ);
      const unsigned bits = __float_as_uint(v);
      if ((bits >> 16) >= (unsigned)thr[b]) {
        unsigned p = atomicAdd(&ccnt[b], 1u);
        if (p < CAP) {
          unsigned flat = (unsigned)(i - b * (NCLS * HWSZ));
          cand[(size_t)b * CAP + p] =
              ((unsigned long long)bits << 32) | (unsigned)(~flat);
        }
      }
    }
  }
}

/* ---------------- exact top-200 + small outputs ---------------- */
__global__ __launch_bounds__(1024) void topk_kernel(
    unsigned long long* __restrict__ cand, const unsigned* __restrict__ ccnt,
    const float* __restrict__ heat_s, float* __restrict__ out,
    int* __restrict__ selflat) {
  __shared__ unsigned long long sk[1024];
  __shared__ int sp[1024];
  __shared__ unsigned long long sel[NPROP];
  const int b = blockIdx.x;
  const int tid = threadIdx.x;
  const unsigned M = min(ccnt[b], (unsigned)CAP);
  unsigned long long* cb = cand + (size_t)b * CAP;

  for (int k = 0; k < NPROP; ++k) {
    unsigned long long best = 0ull;
    int bpos = -1;
    for (unsigned t = tid; t < M; t += 1024) {
      unsigned long long v = cb[t];
      if (v > best) { best = v; bpos = (int)t; }
    }
    sk[tid] = best;
    sp[tid] = bpos;
    __syncthreads();
    for (int s = 512; s > 0; s >>= 1) {
      if (tid < s && sk[tid + s] > sk[tid]) { sk[tid] = sk[tid + s]; sp[tid] = sp[tid + s]; }
      __syncthreads();
    }
    if (tid == 0) {
      sel[k] = sk[0];
      if (sp[0] >= 0) cb[sp[0]] = 0ull;
    }
    __syncthreads();
  }

  if (tid < NPROP) {
    const unsigned long long key = sel[tid];
    unsigned flat = (key == 0ull) ? (unsigned)tid : ~(unsigned)(key & 0xFFFFFFFFu);
    if (flat >= (unsigned)(NCLS * HWSZ)) flat = 0;
    const int clsv = (int)(flat / HWSZ);
    const int idx = (int)(flat % HWSZ);
    out[OFF_CLS + b * NPROP + tid] = (float)clsv;
    out[OFF_POS + (b * NPROP + tid) * 2 + 0] = (float)(idx % WW);
    out[OFF_POS + (b * NPROP + tid) * 2 + 1] = (float)(idx / WW);
#pragma unroll
    for (int c = 0; c < NCLS; ++c)
      out[OFF_QHS + ((b * NCLS) + c) * NPROP + tid] =
          heat_s[(size_t)(b * NCLS + c) * HWSZ + idx];
    selflat[b * NPROP + tid] = (int)flat;
  }
}

/* ---------------- qf gather ---------------- */
__global__ __launch_bounds__(128) void qf_kernel(
    const float* __restrict__ F, const float* __restrict__ Wc,
    const float* __restrict__ bc, const int* __restrict__ p2r,
    const int* __restrict__ selflat, float* __restrict__ out) {
  const int p = blockIdx.x;       /* 0 .. B*NPROP-1 */
  const int b = p / NPROP;
  const int k = p - b * NPROP;
  const int c = threadIdx.x;
  const int flat = selflat[p];
  const int clsv = flat / HWSZ;
  const int idx = flat % HWSZ;
  const int r = p2r[(size_t)b * HWSZ + idx];
  float v = (r >= 0) ? F[(size_t)r * CCH + c] : 0.0f;
  v += Wc[c * NCLS + clsv] + bc[c];
  out[OFF_QF + ((b * CCH) + c) * NPROP + k] = v;
}

extern "C" void kernel_launch(void* const* d_in, const int* in_sizes, int n_in,
                              void* d_out, int out_size, void* d_ws, size_t ws_size,
                              hipStream_t stream) {
  (void)in_sizes; (void)n_in; (void)out_size; (void)ws_size;
  const float* F  = (const float*)d_in[0];
  const float* W1 = (const float*)d_in[1];
  const float* b1 = (const float*)d_in[2];
  const float* W2 = (const float*)d_in[3];
  const float* b2 = (const float*)d_in[4];
  const float* Wc = (const float*)d_in[5];
  const float* bc = (const float*)d_in[6];
  const int* ind  = (const int*)d_in[7];
  float* out = (float*)d_out;
  char* ws = (char*)d_ws;

  float* heat   = (float*)(ws + WS_HEAT);
  float* heat_s = (float*)(ws + WS_HEATS);
  int* p2r      = (int*)(ws + WS_P2R);
  unsigned* hist = (unsigned*)(ws + WS_HIST);
  int* thr      = (int*)(ws + WS_THR);
  unsigned* ccnt = (unsigned*)(ws + WS_CCNT);
  int* selflat  = (int*)(ws + WS_SELF);
  unsigned long long* cand = (unsigned long long*)(ws + WS_CAND);

  hipMemsetAsync(heat, 0, (size_t)BB * NCLS * HWSZ * 4, stream);
  hipMemsetAsync(p2r, 0xFF, (size_t)BB * HWSZ * 4, stream);           /* -1 */
  hipMemsetAsync(ws + WS_HIST, 0, WS_SELF - WS_HIST, stream);          /* hist+thr+ccnt */

  build_map_kernel<<<(NROWS + 255) / 256, 256, 0, stream>>>(ind, p2r);
  head_gemm_kernel<<<GEMM_BLOCKS, 128, 0, stream>>>(F, W1, b1, W2, b2, ind, heat);
  nms_hist_kernel<<<2048, 256, 0, stream>>>(heat, heat_s, hist);
  select_thr_kernel<<<BB, 256, 0, stream>>>(hist, thr);
  compact_kernel<<<2048, 256, 0, stream>>>(heat_s, thr, ccnt, cand);
  topk_kernel<<<BB, 1024, 0, stream>>>(cand, ccnt, heat_s, out, selflat);
  qf_kernel<<<BB * NPROP, 128, 0, stream>>>(F, Wc, bc, p2r, selflat, out);
}
